// Cross_SNR_Loss_53961969107022
// MI455X (gfx1250) — matrix operand order repaired and checked by reading, not compiled, not run on hardware
//
#include <hip/hip_runtime.h>

// CDNA5 / gfx1250 WMMA operand types (probe-verified signatures)
typedef __attribute__((ext_vector_type(16))) _Float16 v16h;
typedef __attribute__((ext_vector_type(8)))  float    v8f;

#define T_LEN   300
#define T_PAD   320          // 10 chunks of K=32
#define F_BINS  210
#define F_PAD   224          // 14 tiles of M=16
#define LOW_BIN 40
#define N_WAVES 14
#define BLOCK   (N_WAVES * 32)
#define TWO_PI_F 6.2831853f  // 2 * 3.14159265 (reference's PI)

__global__ __launch_bounds__(BLOCK)
void csnr_batch_kernel(const float* __restrict__ wave,      // [B, 300]
                       const float* __restrict__ gt_bpm,    // [B]
                       const float* __restrict__ fps,       // [B]
                       const float* __restrict__ hanning,   // [300]
                       float* __restrict__ loss_part,       // [B] scratch
                       float* __restrict__ out_idx)         // [B] -> d_out+1
{
    __shared__ _Float16 sPred[T_PAD];   // Hann-windowed wave, f16, zero-padded
    __shared__ float    sCA[F_PAD];     // complex_absolute per freq bin

    const int b    = blockIdx.x;
    const int tid  = threadIdx.x;
    const int lane = tid & 31;
    const int wv   = tid >> 5;          // 0..13 : freq tile

    // ---- stage preds = wave * hanning into LDS (f16 for WMMA B operand) ----
    for (int t = tid; t < T_PAD; t += BLOCK) {
        float p = (t < T_LEN) ? wave[b * T_LEN + t] * hanning[t] : 0.0f;
        sPred[t] = (_Float16)p;
    }
    __syncthreads();

    const float fps_b = fps[b];

    // WMMA 16-bit A layout (16x32, MxK): lane = khalf*16 + m;
    // element e of the v16h maps to K = 8*khalf + (e&7) + 16*(e>>3).
    const int   m      = lane & 15;
    const int   khalf  = lane >> 4;
    const int   f_mine = 16 * wv + m;                       // freq bin (row M)
    const float theta  = TWO_PI_F * ((float)(LOW_BIN + f_mine) / 60.0f) / fps_b;

    // Oscillator bank: sin/cos(theta * t) at this lane's 16 K slots;
    // advanced by a Delta = 32*theta complex rotation each K-chunk.
    float osc_s[16], osc_c[16];
#pragma unroll
    for (int e = 0; e < 16; ++e) {
        const int   K  = 8 * khalf + (e & 7) + ((e >> 3) << 4);
        const float a0 = theta * (float)K;
        osc_s[e] = __sinf(a0);
        osc_c[e] = __cosf(a0);
    }
    const float rot_s = __sinf(theta * 32.0f);
    const float rot_c = __cosf(theta * 32.0f);

    v8f accS = {};   // Sigma preds[t]*sin : f32 accumulators (C/D matrix)
    v8f accC = {};   // Sigma preds[t]*cos
    // B layout (32x16, KxN, f16): element e maps to K = 8*khalf + (e&7) + 16*(e>>3).
    // We replicate preds across all 16 N columns -> column index is don't-care.
    const int bbase = khalf * 8;

#pragma unroll
    for (int ch = 0; ch < T_PAD / 32; ++ch) {
        v16h a_sin, a_cos, bm;
#pragma unroll
        for (int e = 0; e < 16; ++e) {
            a_sin[e] = (_Float16)osc_s[e];
            a_cos[e] = (_Float16)osc_c[e];
            bm[e]    = sPred[ch * 32 + bbase + (e & 7) + ((e >> 3) << 4)];
        }
        accS = __builtin_amdgcn_wmma_f32_16x16x32_f16(
                   false, a_sin, false, bm, (short)0, accS, false, false);
        accC = __builtin_amdgcn_wmma_f32_16x16x32_f16(
                   false, a_cos, false, bm, (short)0, accC, false, false);
        // rotate oscillators forward by 32 time steps
#pragma unroll
        for (int e = 0; e < 16; ++e) {
            const float ns = osc_s[e] * rot_c + osc_c[e] * rot_s;
            osc_c[e]       = osc_c[e] * rot_c - osc_s[e] * rot_s;
            osc_s[e]       = ns;
        }
    }

    // D layout: lane 0 holds M=0..7 (VGPR r -> M=r, N=0); lane 16 holds M=8..15.
    // All N columns are identical (replicated B), so lanes 0/16 cover the tile.
    if (m == 0) {
#pragma unroll
        for (int r = 0; r < 8; ++r) {
            const int   f  = 16 * wv + 8 * khalf + r;
            const float sv = accS[r], cv = accC[r];
            sCA[f] = (f < F_BINS) ? (sv * sv + cv * cv) : 0.0f;
        }
    }
    __syncthreads();

    // ---- per-batch epilogue on wave 0 (wave32 shuffle reductions) ----
    if (wv == 0) {
        float sum = 0.0f, mx = -1.0f;
        int   mi  = 0x7fffffff;
        for (int f = lane; f < F_BINS; f += 32) {
            const float v = sCA[f];
            sum += v;
            if (v > mx) { mx = v; mi = f; }     // keeps first max within lane
        }
#pragma unroll
        for (int off = 16; off > 0; off >>= 1) sum += __shfl_xor(sum, off, 32);
#pragma unroll
        for (int off = 16; off > 0; off >>= 1) {
            const float omx = __shfl_xor(mx, off, 32);
            const int   omi = __shfl_xor(mi, off, 32);
            if (omx > mx || (omx == mx && omi < mi)) { mx = omx; mi = omi; }
        }
        const float inv   = 1.0f / sum;         // ca_n = ca * inv
        const float mnorm = mx * inv;           // max of ca_n
        float se = 0.0f;
        for (int f = lane; f < F_BINS; f += 32) se += __expf(sCA[f] * inv - mnorm);
#pragma unroll
        for (int off = 16; off > 0; off >>= 1) se += __shfl_xor(se, off, 32);

        if (lane == 0) {
            // target bin
            float hr = gt_bpm[b] * fps_b * 60.0f / 300.0f;
            hr = fminf(fmaxf(hr, (float)LOW_BIN), (float)(LOW_BIN + F_BINS - 1));
            const int target = (int)hr - LOW_BIN;
            // cross-entropy term: -log_softmax(ca_n)[target]
            const float lse = mnorm + __logf(se);
            const float ce  = -(sCA[target] * inv - lse);
            // SNR window [target-3, target+3) clipped
            int l = target - 3; if (l < 0) l = 0;
            int r = target + 3; if (r > F_BINS - 1) r = F_BINS - 1;
            float wsum = 0.0f;
            for (int f = l; f < r; ++f) wsum += sCA[f] * inv;

            loss_part[b] = ce + (1.0f - wsum);
            out_idx[b]   = (float)(mi + LOW_BIN);
        }
    }
}

__global__ __launch_bounds__(1024)
void csnr_finalize_kernel(const float* __restrict__ part, float* __restrict__ out, int B)
{
    __shared__ float sm[1024];
    const int t = threadIdx.x;
    float v = 0.0f;
    for (int i = t; i < B; i += 1024) v += part[i];
    sm[t] = v;
    __syncthreads();
    for (int s = 512; s > 0; s >>= 1) {
        if (t < s) sm[t] += sm[t + s];
        __syncthreads();
    }
    if (t == 0) out[0] = sm[0] / (float)B;
}

extern "C" void kernel_launch(void* const* d_in, const int* in_sizes, int n_in,
                              void* d_out, int out_size, void* d_ws, size_t ws_size,
                              hipStream_t stream)
{
    const float* wave    = (const float*)d_in[0];
    const float* gt_bpm  = (const float*)d_in[1];
    const float* fps     = (const float*)d_in[2];
    const float* hanning = (const float*)d_in[3];
    float* out  = (float*)d_out;        // out[0] = loss; out[1..B] = argmax+40
    float* part = (float*)d_ws;         // B floats of per-batch loss
    const int B = in_sizes[1];          // gt_bpm is [B]

    csnr_batch_kernel<<<B, BLOCK, 0, stream>>>(wave, gt_bpm, fps, hanning,
                                               part, out + 1);
    csnr_finalize_kernel<<<1, 1024, 0, stream>>>(part, out, B);
}
